// ContextAttention_30270929502708
// MI455X (gfx1250) — matrix operand order repaired and checked by reading, not compiled, not run on hardware
//
#include <hip/hip_runtime.h>
#include <hip/hip_bf16.h>
#include <float.h>

typedef __bf16 bf16_t;
typedef __attribute__((ext_vector_type(16))) __bf16 v16bf;
typedef __attribute__((ext_vector_type(8)))  float  v8f;

#define NB  2
#define CCH 64
#define MM  4800
#define KK  64
#define BNEPS 1e-5f

// ---------------------------------------------------------------------------
// Kernel 1: per-(n,k) precompute.
//   A0s[n,k,c] = s0[c] * (W0[:, :64] @ scene_rgb[n,:,k])          (bn0 scale folded)
//   B1 [n,k,c] = s1a[c]*(W1a[:,64:67] @ (xyz*mask)) + t1a[c]      (bn1a folded)
//   BS [n,k,c] = Wskip[:,64:67] @ (xyz*mask)
// ---------------------------------------------------------------------------
__global__ void prep_k_kernel(
    const float* __restrict__ scene_rgb, const float* __restrict__ scene_xyz,
    const float* __restrict__ mask, const float* __restrict__ W0,
    const float* __restrict__ g0, const float* __restrict__ v0,
    const float* __restrict__ W1a,
    const float* __restrict__ g1a, const float* __restrict__ b1a,
    const float* __restrict__ m1a, const float* __restrict__ v1a,
    const float* __restrict__ Wskip,
    float* __restrict__ A0s, float* __restrict__ B1, float* __restrict__ BS)
{
    int nk = blockIdx.x;
    int n = nk >> 6, k = nk & 63;
    int c = threadIdx.x;

    float s0 = g0[c] * rsqrtf(v0[c] + BNEPS);
    const float* w0row = W0 + c * (2 * CCH);
    const float* sr = scene_rgb + n * CCH * KK + k;
    float a = 0.f;
    #pragma unroll 8
    for (int cc = 0; cc < CCH; ++cc) a += w0row[cc] * sr[cc * KK];
    A0s[nk * CCH + c] = s0 * a;

    float mk  = mask[n * KK + k];
    float sx0 = scene_xyz[(n * 3 + 0) * KK + k] * mk;
    float sx1 = scene_xyz[(n * 3 + 1) * KK + k] * mk;
    float sx2 = scene_xyz[(n * 3 + 2) * KK + k] * mk;

    float s1 = g1a[c] * rsqrtf(v1a[c] + BNEPS);
    float t1 = b1a[c] - m1a[c] * s1;
    const float* w1row = W1a + c * (CCH + 3) + CCH;
    B1[nk * CCH + c] = s1 * (w1row[0] * sx0 + w1row[1] * sx1 + w1row[2] * sx2) + t1;

    const float* wsrow = Wskip + c * (CCH + 3) + CCH;
    BS[nk * CCH + c] = wsrow[0] * sx0 + wsrow[1] * sx1 + wsrow[2] * sx2;
}

// ---------------------------------------------------------------------------
// Kernel 2: per-(n,m) precompute.
//   B0s[n,m,c] = s0[c] * (W0[:, 64:] @ query[n,:,m]) + t0[c]
// ---------------------------------------------------------------------------
__global__ void prep_m_kernel(
    const float* __restrict__ q, const float* __restrict__ W0,
    const float* __restrict__ g0, const float* __restrict__ b0,
    const float* __restrict__ m0, const float* __restrict__ v0,
    float* __restrict__ B0s)
{
    int m = blockIdx.x, n = blockIdx.y, c = threadIdx.x;
    float s0 = g0[c] * rsqrtf(v0[c] + BNEPS);
    float t0 = b0[c] - m0[c] * s0;
    const float* w  = W0 + c * (2 * CCH) + CCH;
    const float* qc = q + n * CCH * MM + m;
    float a = 0.f;
    #pragma unroll 8
    for (int cc = 0; cc < CCH; ++cc) a += w[cc] * qc[cc * MM];
    B0s[((size_t)(n * MM) + m) * CCH + c] = s0 * a + t0;
}

// ---------------------------------------------------------------------------
// Kernel 3: fused main pipeline. One m per wave, 4 k-tiles of 16 points.
// bf16 WMMA (f32 accumulate) for the three 64x64 GEMMs; BN folded.
// ---------------------------------------------------------------------------
__device__ __forceinline__ v16bf ld_frag(const bf16_t* p) {
    return *(const v16bf*)p;   // two ds_load_b128 (32B aligned)
}

__device__ __forceinline__ v8f wmma_bf16(v16bf a, v16bf b, v8f c) {
    return __builtin_amdgcn_wmma_f32_16x16x32_bf16(false, a, false, b,
                                                   (short)0, c, false, false);
}

__launch_bounds__(256)
__global__ void main_kernel(
    const float* __restrict__ A0g, const float* __restrict__ B1g,
    const float* __restrict__ BSg, const float* __restrict__ B0s,
    const float* __restrict__ W1a, const float* __restrict__ W1b,
    const float* __restrict__ Wskip,
    const float* __restrict__ g1a, const float* __restrict__ v1a,
    const float* __restrict__ g1b, const float* __restrict__ b1b,
    const float* __restrict__ m1b, const float* __restrict__ v1b,
    const float* __restrict__ Wout, const float* __restrict__ pre_xyz,
    float* __restrict__ out)
{
    // k-indexed slabs padded to stride 65 (bank-conflict-free column reads)
    __shared__ __align__(32) float  A0l[64 * 65];
    __shared__ __align__(32) float  B1l[64 * 65];
    __shared__ __align__(32) float  BSl[64 * 65];
    // weights pre-swizzled into exact 16-bit 16x32 A-fragment order:
    // idx = ((rt*2 + j)*32 + lane)*16 + e
    __shared__ __align__(32) bf16_t Wfrag[3 * 4096];
    // per-wave t1 round-trip buffer, [p*80 + c] (stride 80 -> 32B aligned frags)
    __shared__ __align__(32) bf16_t T1[8 * 16 * 80];
    __shared__ float FM[8][64];
    __shared__ float T1Bl[64];

    int tid = threadIdx.x;
    int n   = blockIdx.y;

    for (int i = tid; i < 64 * 64; i += 256) {
        int k = i >> 6, c = i & 63;
        A0l[k * 65 + c] = A0g[n * 4096 + i];
        B1l[k * 65 + c] = B1g[n * 4096 + i];
        BSl[k * 65 + c] = BSg[n * 4096 + i];
    }
    for (int i = tid; i < 4096; i += 256) {
        int e  = i & 15;
        int L  = (i >> 4) & 31;
        int rj = i >> 9;              // rt*2 + j
        int rt = rj >> 1, j = rj & 1;
        // 16-bit A-matrix 16x32, contiguous k order: k = 16*(L/16) + e
        int kk = e + ((L < 16) ? 0 : 16);
        int ci = j * 32 + kk;
        int co = rt * 16 + (L & 15);
        float s1 = g1a[co] * rsqrtf(v1a[co] + BNEPS);
        Wfrag[i]        = (bf16_t)(s1 * W1a[co * 67 + ci]);   // bn1a scale folded
        float s2 = g1b[co] * rsqrtf(v1b[co] + BNEPS);
        Wfrag[4096 + i] = (bf16_t)(s2 * W1b[co * 64 + ci]);   // bn1b scale folded
        Wfrag[8192 + i] = (bf16_t)(Wskip[co * 67 + ci]);
    }
    if (tid < 64) {
        float s2 = g1b[tid] * rsqrtf(v1b[tid] + BNEPS);
        T1Bl[tid] = b1b[tid] - m1b[tid] * s2;
    }
    __syncthreads();

    int wave = tid >> 5, lane = tid & 31;
    int p = lane & 15, hi = lane >> 4;
    int m = blockIdx.x * 8 + wave;

    // per-m fcn0 half, loaded into the B-fragment channel order (broadcast loads)
    const float* B0row = B0s + ((size_t)(n * MM) + m) * 64;
    float bq[2][16];
    #pragma unroll
    for (int j = 0; j < 2; ++j) {
        const float4* src = (const float4*)(B0row + j * 32 + hi * 16);
        #pragma unroll
        for (int q4 = 0; q4 < 4; ++q4) {
            float4 vv = src[q4];
            bq[j][q4 * 4 + 0] = vv.x; bq[j][q4 * 4 + 1] = vv.y;
            bq[j][q4 * 4 + 2] = vv.z; bq[j][q4 * 4 + 3] = vv.w;
        }
    }

    float vmax[4][8];
    #pragma unroll
    for (int rt = 0; rt < 4; ++rt)
        #pragma unroll
        for (int v = 0; v < 8; ++v) vmax[rt][v] = -FLT_MAX;

    bf16_t* T1w = &T1[wave * (16 * 80)];

    for (int kt = 0; kt < 4; ++kt) {
        int k = kt * 16 + p;

        // h = relu(A0s[k] + B0s[m]) in bf16 B-fragment layout (K split 2x32)
        v16bf hB0, hB1;
        #pragma unroll
        for (int e = 0; e < 16; ++e) {
            int c0 = hi * 16 + e;
            float h0 = A0l[k * 65 + c0]      + bq[0][e];
            float h1 = A0l[k * 65 + 32 + c0] + bq[1][e];
            hB0[e] = (bf16_t)fmaxf(h0, 0.f);
            hB1[e] = (bf16_t)fmaxf(h1, 0.f);
        }

        // t1 = relu(W1a' @ h + B1[k]); re-fragment via per-wave LDS transpose
        #pragma unroll
        for (int rt = 0; rt < 4; ++rt) {
            v8f acc = {0.f, 0.f, 0.f, 0.f, 0.f, 0.f, 0.f, 0.f};
            acc = wmma_bf16(ld_frag(&Wfrag[((rt * 2 + 0) * 32 + lane) * 16]), hB0, acc);
            acc = wmma_bf16(ld_frag(&Wfrag[((rt * 2 + 1) * 32 + lane) * 16]), hB1, acc);
            #pragma unroll
            for (int v = 0; v < 8; ++v) {
                int co = rt * 16 + hi * 8 + v;         // C/D layout row
                float t = fmaxf(acc[v] + B1l[k * 65 + co], 0.f);
                T1w[p * 80 + co] = (bf16_t)t;
            }
        }
        // same-wave LDS ops are in-order -> safe to reload as B-fragments
        v16bf tB0 = ld_frag(&T1w[p * 80 + hi * 16]);
        v16bf tB1 = ld_frag(&T1w[p * 80 + 32 + hi * 16]);

        #pragma unroll
        for (int rt = 0; rt < 4; ++rt) {
            v8f a2 = {0.f, 0.f, 0.f, 0.f, 0.f, 0.f, 0.f, 0.f};
            a2 = wmma_bf16(ld_frag(&Wfrag[4096 + ((rt * 2 + 0) * 32 + lane) * 16]), tB0, a2);
            a2 = wmma_bf16(ld_frag(&Wfrag[4096 + ((rt * 2 + 1) * 32 + lane) * 16]), tB1, a2);
            v8f a3 = {0.f, 0.f, 0.f, 0.f, 0.f, 0.f, 0.f, 0.f};
            a3 = wmma_bf16(ld_frag(&Wfrag[8192 + ((rt * 2 + 0) * 32 + lane) * 16]), hB0, a3);
            a3 = wmma_bf16(ld_frag(&Wfrag[8192 + ((rt * 2 + 1) * 32 + lane) * 16]), hB1, a3);
            #pragma unroll
            for (int v = 0; v < 8; ++v) {
                int co = rt * 16 + hi * 8 + v;
                float t2 = fmaxf(a2[v] + T1Bl[co], 0.f);
                float f  = t2 + a3[v] + BSl[k * 65 + co];
                vmax[rt][v] = fmaxf(vmax[rt][v], f);
            }
        }
    }

    // max over the 16 points of each half-wave group (k covered across kt already)
    #pragma unroll
    for (int rt = 0; rt < 4; ++rt)
        #pragma unroll
        for (int v = 0; v < 8; ++v) {
            float x = vmax[rt][v];
            #pragma unroll
            for (int off = 1; off < 16; off <<= 1)
                x = fmaxf(x, __shfl_xor(x, off, 32));
            if (p == 0) FM[wave][rt * 16 + hi * 8 + v] = x;
        }

    // output head: out[n,o,m] = Wout[o,:64]@featmax + Wout[o,64:]@pre_xyz[n,:,m]
    if (lane < 3) {
        float acc = 0.f;
        const float* wo = Wout + lane * 67;
        #pragma unroll 8
        for (int c = 0; c < 64; ++c) acc += wo[c] * FM[wave][c];
        #pragma unroll
        for (int i = 0; i < 3; ++i) acc += wo[64 + i] * pre_xyz[(n * 3 + i) * MM + m];
        out[(n * 3 + lane) * MM + m] = acc;
    }
}

// ---------------------------------------------------------------------------
extern "C" void kernel_launch(void* const* d_in, const int* in_sizes, int n_in,
                              void* d_out, int out_size, void* d_ws, size_t ws_size,
                              hipStream_t stream) {
    const float* q     = (const float*)d_in[0];
    const float* srgb  = (const float*)d_in[1];
    const float* sxyz  = (const float*)d_in[2];
    const float* pxyz  = (const float*)d_in[3];
    const float* mask  = (const float*)d_in[4];
    const float* W0    = (const float*)d_in[5];
    const float* g0    = (const float*)d_in[6];
    const float* b0    = (const float*)d_in[7];
    const float* m0    = (const float*)d_in[8];
    const float* v0    = (const float*)d_in[9];
    const float* W1a   = (const float*)d_in[10];
    const float* g1a   = (const float*)d_in[11];
    const float* b1a   = (const float*)d_in[12];
    const float* m1a   = (const float*)d_in[13];
    const float* v1a   = (const float*)d_in[14];
    const float* W1b   = (const float*)d_in[15];
    const float* g1b   = (const float*)d_in[16];
    const float* b1b   = (const float*)d_in[17];
    const float* m1b   = (const float*)d_in[18];
    const float* v1b   = (const float*)d_in[19];
    const float* Wskip = (const float*)d_in[20];
    const float* Wout  = (const float*)d_in[21];
    float* out = (float*)d_out;

    float* ws  = (float*)d_ws;
    float* A0s = ws;                 // N*K*C = 8192 floats
    float* B1  = ws + 8192;
    float* BS  = ws + 16384;
    float* B0s = ws + 24576;         // N*M*C = 614400 floats

    prep_k_kernel<<<dim3(NB * KK), dim3(64), 0, stream>>>(
        srgb, sxyz, mask, W0, g0, v0, W1a, g1a, b1a, m1a, v1a, Wskip,
        A0s, B1, BS);
    prep_m_kernel<<<dim3(MM, NB), dim3(64), 0, stream>>>(
        q, W0, g0, b0, m0, v0, B0s);
    main_kernel<<<dim3(MM / 8, NB), dim3(256), 0, stream>>>(
        A0s, B1, BS, B0s, W1a, W1b, Wskip, g1a, v1a, g1b, b1b, m1b, v1b,
        Wout, pxyz, out);
}
